// GraphReasoning_21397527069096
// MI455X (gfx1250) — matrix operand order repaired and checked by reading, not compiled, not run on hardware
//
#include <hip/hip_runtime.h>
#include <hip/hip_bf16.h>

typedef _Float16 v16h __attribute__((ext_vector_type(16)));
typedef _Float16 v8h  __attribute__((ext_vector_type(8)));
typedef float    v8f  __attribute__((ext_vector_type(8)));
typedef float    v4f  __attribute__((ext_vector_type(4)));

// Problem constants: B=8, C=256, H=W=64, N_NODE=4, N_ITER=2
// d_out = [x: 8*512*64*64 floats][sm: 8*5*64*64 floats]
static constexpr size_t X_ELEMS  = 8ull*512*64*64;      // 16,777,216

// Workspace layout (bytes, all 256B aligned)
static constexpr size_t IN_H_OFF  = 0;                  // f16 [8][66][66][256]  17,842,176 B
static constexpr size_t WB1_OFF   = 17842176;           // f16 [9][16][256]         73,728 B
static constexpr size_t FEATA_OFF = 17915904;           // f32 [8][4][256]          32,768 B
static constexpr size_t FEATB_OFF = 17948672;           // f32 [8][4][256]          32,768 B
static constexpr size_t WB2_OFF   = 17981440;           // f16 [2304][1024]      4,718,592 B
static constexpr size_t R_OFF     = 22700032;           // f32 [8][2304]            73,728 B
static constexpr size_t S_OFF     = 22773760;           // f32 [8][256][9]          73,728 B
static constexpr size_t WBG_OFF   = 22847488;           // f16 [512][512]          524,288 B
static constexpr size_t WBC_OFF   = 23371776;           // f16 [256][512]          262,144 B
static constexpr size_t XC_OFF    = 23633920;           // f16 [32][512]            32,768 B
static constexpr size_t XC2_OFF   = 23666688;           // f16 [32][512]            32,768 B
static constexpr size_t U_OFF     = 23699456;           // f32 [32][256]            32,768 B

__device__ __forceinline__ float sigf(float x) { return 1.0f / (1.0f + __expf(-x)); }

// ---------------------------------------------------------------------------
// K0: inputs f32 NCHW -> zero-padded NHWC f16  [b][66][66][256] (LDS transpose)
// grid = 8*64*4, block = 256
__global__ void k_in2h(const float* __restrict__ in, _Float16* __restrict__ inh) {
  int blk = blockIdx.x;
  int b = blk >> 8, rem = blk & 255;
  int y = rem >> 2, x0 = (rem & 3) << 4;
  int t = threadIdx.x;
  __shared__ float tile[256][17];
  int xi = t & 15, cb = t >> 4;
  #pragma unroll
  for (int it = 0; it < 16; ++it) {
    int c = it * 16 + cb;
    tile[c][xi] = in[((b * 256 + c) * 64 + y) * 64 + x0 + xi];
  }
  __syncthreads();
  #pragma unroll
  for (int x = 0; x < 16; ++x)
    inh[(((size_t)(b * 66 + (y + 1)) * 66) + (x0 + x + 1)) * 256 + t] = (_Float16)tile[t][x];
}

// K0b: zero the padded border cells. grid = 8*260, block = 256
__global__ void k_hborder(_Float16* __restrict__ inh) {
  int blk = blockIdx.x;
  int b = blk / 260, e = blk % 260, t = threadIdx.x;
  int py, px;
  if (e < 132) { py = (e < 66) ? 0 : 65; px = (e < 66) ? e : e - 66; }
  else         { int r = e - 132; px = (r < 64) ? 0 : 65; py = (r < 64) ? r + 1 : r - 64 + 1; }
  inh[(((size_t)(b * 66 + py) * 66) + px) * 256 + t] = (_Float16)0.0f;
}

// K1: fc_w (5,256,3,3) f32 -> wb1 [tap][n(16, pad0)][c] f16. grid = 144, block = 256
__global__ void k_wb1(const float* __restrict__ fcw, _Float16* __restrict__ wb1) {
  int e = blockIdx.x * 256 + threadIdx.x;   // tap*4096 + n*256 + c
  int tap = e >> 12, rem = e & 4095, n = rem >> 8, c = rem & 255;
  float v = 0.0f;
  if (n < 5) v = fcw[((n * 256 + c) * 3 + tap / 3) * 3 + (tap % 3)];
  wb1[e] = (_Float16)v;
}

// K1b: plain f32 -> f16 convert (used for gru_gw, gru_cw). grid = n/256, block = 256
__global__ void k_cvt(const float* __restrict__ src, _Float16* __restrict__ dst) {
  size_t e = (size_t)blockIdx.x * 256 + threadIdx.x;
  dst[e] = (_Float16)src[e];
}

// K2: conv1 (implicit GEMM via v_wmma_f32_16x16x32_f16) + bias + softmax -> sm
// grid = 8*64*4 waves (tile = 16 pixels of one row), block = 32 (one wave)
__global__ void k_conv1(const _Float16* __restrict__ inh, const _Float16* __restrict__ wb1,
                        const float* __restrict__ fcb, float* __restrict__ sm) {
  int blk = blockIdx.x;
  int b = blk >> 8, rem = blk & 255;
  int y = rem >> 2, x0 = (rem & 3) << 4;
  int lane = threadIdx.x;
  int nl = lane & 15, half = lane >> 4;
  v8f acc = {};
  for (int tap = 0; tap < 9; ++tap) {
    int dy = tap / 3, dx = tap % 3;
    const _Float16* rp = inh + (((size_t)(b * 66 + y + dy) * 66) + (x0 + nl + dx)) * 256;
    const _Float16* wp = wb1 + ((size_t)tap * 16 + nl) * 256;
    #pragma unroll
    for (int cb = 0; cb < 8; ++cb) {
      int c0 = cb * 32;
      v8h alo = *(const v8h*)(rp + c0 + half * 16);       // K = c0 + half*16 + 0..7
      v8h ahi = *(const v8h*)(rp + c0 + half * 16 + 8);   // K = c0 + half*16 + 8..15
      v16h a = __builtin_shufflevector(alo, ahi, 0,1,2,3,4,5,6,7,8,9,10,11,12,13,14,15);
      v16h bm = *(const v16h*)(wp + c0 + half * 16);      // K = c0 + half*16 + e
      acc = __builtin_amdgcn_wmma_f32_16x16x32_f16(false, a, false, bm, (short)0, acc, false, false);
    }
  }
  __shared__ float sml[16][17];                            // [pixel][out-channel]
  #pragma unroll
  for (int r = 0; r < 8; ++r) sml[r + 8 * half][nl] = acc[r];
  __syncthreads();
  if (lane < 16) {
    int x = x0 + lane;
    float v[5], mx = -1e30f;
    #pragma unroll
    for (int k = 0; k < 5; ++k) { v[k] = sml[lane][k] + fcb[k]; mx = fmaxf(mx, v[k]); }
    float s = 0.0f;
    #pragma unroll
    for (int k = 0; k < 5; ++k) { v[k] = __expf(v[k] - mx); s += v[k]; }
    float inv = 1.0f / s;
    #pragma unroll
    for (int k = 0; k < 5; ++k)
      sm[((size_t)(b * 5 + k) * 64 + y) * 64 + x] = v[k] * inv;
  }
}

// K3: centers[b,k,c] = mean_p sm[b,k+1,p]*in[b,c,p] -> featA. grid = 8*256, block = 256
__global__ void k_centers(const float* __restrict__ in, const float* __restrict__ sm,
                          float* __restrict__ feat) {
  int b = blockIdx.x >> 8, c = blockIdx.x & 255, t = threadIdx.x;
  const float* ip = in + (size_t)(b * 256 + c) * 4096;
  const float* sp = sm + (size_t)b * 5 * 4096;
  float s0 = 0, s1 = 0, s2 = 0, s3 = 0;
  for (int p = t; p < 4096; p += 256) {
    float xv = ip[p];
    s0 += sp[1 * 4096 + p] * xv;
    s1 += sp[2 * 4096 + p] * xv;
    s2 += sp[3 * 4096 + p] * xv;
    s3 += sp[4 * 4096 + p] * xv;
  }
  __shared__ float red[256];
  float ss[4] = {s0, s1, s2, s3};
  for (int k = 0; k < 4; ++k) {
    red[t] = ss[k];
    __syncthreads();
    for (int w = 128; w > 0; w >>= 1) { if (t < w) red[t] += red[t + w]; __syncthreads(); }
    if (t == 0) feat[((size_t)b * 4 + k) * 256 + c] = red[0] * (1.0f / 4096.0f);
    __syncthreads();
  }
}

// K4a: message pass per (b,node): edge weights + m_t; emit Xc=[m_t|h_t] f16 and
// the m_t half of Xc2. grid = 32 (row = b*4+idx), block = 256
__global__ void k_gmsg(const float* __restrict__ src, const float* __restrict__ cwgt,
                       _Float16* __restrict__ Xc, _Float16* __restrict__ Xc2) {
  int b = blockIdx.x >> 2, idx = blockIdx.x & 3, t = threadIdx.x;
  int row = blockIdx.x;
  __shared__ float featS[4][256];
  __shared__ float red[256];
  __shared__ float wgtS[3];
  for (int i = 0; i < 4; ++i) featS[i][t] = src[((size_t)b * 4 + i) * 256 + t];
  __syncthreads();
  int js[3]; { int k = 0; for (int j = 0; j < 4; ++j) if (j != idx) js[k++] = j; }
  for (int g = 0; g < 3; ++g) {                       // grouped 1x1 conv on diff
    red[t] = cwgt[g * 256 + t] * (featS[idx][t] - featS[js[g]][t]);
    __syncthreads();
    for (int w = 128; w > 0; w >>= 1) { if (t < w) red[t] += red[t + w]; __syncthreads(); }
    if (t == 0) wgtS[g] = red[0];
    __syncthreads();
  }
  float sw[3];
  for (int g = 0; g < 3; ++g) sw[g] = sigf(wgtS[g]);
  // reference quirk: gate index (g*256+c)%3 == (g+c)%3
  float m = 0.0f;
  for (int g = 0; g < 3; ++g) m += featS[js[g]][t] * sw[(g + t) % 3];
  Xc [(size_t)row * 512 + t]       = (_Float16)m;
  Xc2[(size_t)row * 512 + t]       = (_Float16)m;
  Xc [(size_t)row * 512 + 256 + t] = (_Float16)featS[idx][t];
}

// K4b: gate GEMM G = Xc(32x512) x gw^T (WMMA) + sigmoid epilogue.
// o<256 -> Xc2[:,256+o] = r*h ; o>=256 -> U[:,o-256] = u.
// grid = 2 mtiles * 32 ntiles = 64, block = 32 (one wave)
__global__ void k_ggemm(const _Float16* __restrict__ Xc, const _Float16* __restrict__ wbg,
                        const float* __restrict__ gb, const float* __restrict__ src,
                        _Float16* __restrict__ Xc2, float* __restrict__ U) {
  int mtile = blockIdx.x >> 5, nt = blockIdx.x & 31;
  int lane = threadIdx.x, nl = lane & 15, half = lane >> 4;
  const _Float16* ap = Xc  + (size_t)(mtile * 16 + nl) * 512;
  const _Float16* bp = wbg + (size_t)(nt * 16 + nl) * 512;
  v8f acc = {};
  #pragma unroll
  for (int kc = 0; kc < 16; ++kc) {
    int k0 = kc * 32;
    v8h alo = *(const v8h*)(ap + k0 + half * 16);
    v8h ahi = *(const v8h*)(ap + k0 + half * 16 + 8);
    v16h a = __builtin_shufflevector(alo, ahi, 0,1,2,3,4,5,6,7,8,9,10,11,12,13,14,15);
    v16h bm = *(const v16h*)(bp + k0 + half * 16);
    acc = __builtin_amdgcn_wmma_f32_16x16x32_f16(false, a, false, bm, (short)0, acc, false, false);
  }
  int o = nt * 16 + nl;
  float bias = gb[o];
  #pragma unroll
  for (int r = 0; r < 8; ++r) {
    int m = mtile * 16 + r + 8 * half;                 // row 0..31, all valid
    float d = sigf(acc[r] + bias);
    if (o < 256) {                                     // wave-uniform branch
      float h = src[(size_t)m * 256 + o];
      Xc2[(size_t)m * 512 + 256 + o] = (_Float16)(d * h);
    } else {
      U[(size_t)m * 256 + (o - 256)] = d;
    }
  }
}

// K4c: candidate GEMM C = Xc2(32x512) x cw^T (WMMA) + tanh + GRU update + gamma residual.
// grid = 2 mtiles * 16 ntiles = 32, block = 32 (one wave)
__global__ void k_cgemm(const _Float16* __restrict__ Xc2, const _Float16* __restrict__ wbc,
                        const float* __restrict__ cbv, const float* __restrict__ src,
                        const float* __restrict__ U, const float* __restrict__ gamma_p,
                        float* __restrict__ dst) {
  int mtile = blockIdx.x >> 4, nt = blockIdx.x & 15;
  int lane = threadIdx.x, nl = lane & 15, half = lane >> 4;
  const _Float16* ap = Xc2 + (size_t)(mtile * 16 + nl) * 512;
  const _Float16* bp = wbc + (size_t)(nt * 16 + nl) * 512;
  v8f acc = {};
  #pragma unroll
  for (int kc = 0; kc < 16; ++kc) {
    int k0 = kc * 32;
    v8h alo = *(const v8h*)(ap + k0 + half * 16);
    v8h ahi = *(const v8h*)(ap + k0 + half * 16 + 8);
    v16h a = __builtin_shufflevector(alo, ahi, 0,1,2,3,4,5,6,7,8,9,10,11,12,13,14,15);
    v16h bm = *(const v16h*)(bp + k0 + half * 16);
    acc = __builtin_amdgcn_wmma_f32_16x16x32_f16(false, a, false, bm, (short)0, acc, false, false);
  }
  int n = nt * 16 + nl;
  float bias = cbv[n];
  float gm = gamma_p[0];
  #pragma unroll
  for (int r = 0; r < 8; ++r) {
    int m = mtile * 16 + r + 8 * half;
    float cand = tanhf(acc[r] + bias);
    float h = src[(size_t)m * 256 + n];
    float u = U[(size_t)m * 256 + n];
    float hn = (1.0f - u) * h + u * cand;
    dst[(size_t)m * 256 + n] = hn * gm + h;
  }
}

// K5: conv_w (256,1024,3,3) f32 -> wb2 [n=o*9+tap][ic] f16. grid = 9216, block = 256
__global__ void k_wb2(const float* __restrict__ cwv, _Float16* __restrict__ wb2) {
  size_t e = (size_t)blockIdx.x * 256 + threadIdx.x;  // n*1024 + ic
  int n = (int)(e >> 10), ic = (int)(e & 1023);
  int o = n / 9, tp = n % 9;
  wb2[e] = (_Float16)cwv[((size_t)o * 1024 + ic) * 9 + tp];
}

// K6: R[b, o*9+tap] = sum_ic feat[b,ic] * conv_w[o,ic,tap]  (WMMA, M=16 pad batch)
// grid = 144 n-tiles, block = 32 (one wave)
__global__ void k_fgemm(const float* __restrict__ feat, const _Float16* __restrict__ wb2,
                        float* __restrict__ R) {
  int nt = blockIdx.x, lane = threadIdx.x;
  int nl = lane & 15, half = lane >> 4;
  int n = nt * 16 + nl;
  const float* ap = feat + (size_t)(nl & 7) * 1024;   // rows >=8 are zero (masked below)
  v8f acc = {};
  for (int kc = 0; kc < 32; ++kc) {
    int k0 = kc * 32;
    v16h a;
    #pragma unroll
    for (int e = 0; e < 16; ++e) {
      int k = k0 + half * 16 + e;
      float fv = ap[k];
      a[e] = (_Float16)((nl < 8) ? fv : 0.0f);
    }
    v16h bm = *(const v16h*)(wb2 + (size_t)n * 1024 + k0 + half * 16);
    acc = __builtin_amdgcn_wmma_f32_16x16x32_f16(false, a, false, bm, (short)0, acc, false, false);
  }
  #pragma unroll
  for (int r = 0; r < 8; ++r) {
    int bb = r + 8 * half;
    if (bb < 8) R[(size_t)bb * 2304 + n] = acc[r];
  }
}

// K7: S[b,o,cat] = conv_b[o] + sum of valid taps of R (9 border categories). grid = 8, block = 256
__global__ void k_compS(const float* __restrict__ R, const float* __restrict__ cbv,
                        float* __restrict__ S) {
  int gid = blockIdx.x * 256 + threadIdx.x;           // 2048 = 8*256
  int b = gid >> 8, o = gid & 255;
  const float* Rp = R + (size_t)b * 2304 + o * 9;
  float bias = cbv[o];
  for (int cy = 0; cy < 3; ++cy) {
    int ky0 = (cy == 0) ? 1 : 0, ky1 = (cy == 2) ? 1 : 2;
    for (int cx = 0; cx < 3; ++cx) {
      int kx0 = (cx == 0) ? 1 : 0, kx1 = (cx == 2) ? 1 : 2;
      float s = bias;
      for (int ky = ky0; ky <= ky1; ++ky)
        for (int kx = kx0; kx <= kx1; ++kx) s += Rp[ky * 3 + kx];
      S[((size_t)(b * 256 + o)) * 9 + cy * 3 + cx] = s;
    }
  }
}

// K8: x = concat(inputs, fused) streamed as float4 with non-temporal stores.
// grid = 16384, block = 256
__global__ void k_out(const float* __restrict__ in, const float* __restrict__ S,
                      float* __restrict__ out) {
  size_t i4 = ((size_t)blockIdx.x * 256 + threadIdx.x) * 4;
  int b = (int)(i4 >> 21);
  int rem = (int)(i4 & 2097151);
  int ch = rem >> 12, p = rem & 4095;
  int y = p >> 6, x = p & 63;
  v4f v;
  if (ch < 256) {
    v = *(const v4f*)(in + (size_t)(b * 256 + ch) * 4096 + p);
  } else {
    const float* Sp = S + ((size_t)(b * 256 + (ch - 256))) * 9;
    int cy = (y == 0) ? 0 : ((y == 63) ? 2 : 1);
    #pragma unroll
    for (int q = 0; q < 4; ++q) {
      int xx = x + q;
      int cx = (xx == 0) ? 0 : ((xx == 63) ? 2 : 1);
      v[q] = Sp[cy * 3 + cx];
    }
  }
  __builtin_nontemporal_store(v, (v4f*)(out + i4));
}

// ---------------------------------------------------------------------------
extern "C" void kernel_launch(void* const* d_in, const int* in_sizes, int n_in,
                              void* d_out, int out_size, void* d_ws, size_t ws_size,
                              hipStream_t stream) {
  const float* inputs = (const float*)d_in[0];
  const float* fc_w   = (const float*)d_in[1];
  const float* fc_b   = (const float*)d_in[2];
  const float* cwgt_w = (const float*)d_in[3];
  const float* gru_gw = (const float*)d_in[4];
  const float* gru_gb = (const float*)d_in[5];
  const float* gru_cw = (const float*)d_in[6];
  const float* gru_cb = (const float*)d_in[7];
  const float* gamma  = (const float*)d_in[8];
  const float* conv_w = (const float*)d_in[9];
  const float* conv_b = (const float*)d_in[10];

  float* out = (float*)d_out;
  float* sm  = out + X_ELEMS;

  char* ws = (char*)d_ws;
  _Float16* in_h  = (_Float16*)(ws + IN_H_OFF);
  _Float16* wb1   = (_Float16*)(ws + WB1_OFF);
  float*    featA = (float*)(ws + FEATA_OFF);
  float*    featB = (float*)(ws + FEATB_OFF);
  _Float16* wb2   = (_Float16*)(ws + WB2_OFF);
  float*    Rbuf  = (float*)(ws + R_OFF);
  float*    Sbuf  = (float*)(ws + S_OFF);
  _Float16* wbg   = (_Float16*)(ws + WBG_OFF);
  _Float16* wbc   = (_Float16*)(ws + WBC_OFF);
  _Float16* Xc    = (_Float16*)(ws + XC_OFF);
  _Float16* Xc2   = (_Float16*)(ws + XC2_OFF);
  float*    Ubuf  = (float*)(ws + U_OFF);

  // prep
  k_in2h   <<<2048, 256, 0, stream>>>(inputs, in_h);
  k_hborder<<<2080, 256, 0, stream>>>(in_h);
  k_wb1    <<<144,  256, 0, stream>>>(fc_w, wb1);
  k_wb2    <<<9216, 256, 0, stream>>>(conv_w, wb2);
  k_cvt    <<<1024, 256, 0, stream>>>(gru_gw, wbg);   // 512*512
  k_cvt    <<<512,  256, 0, stream>>>(gru_cw, wbc);   // 256*512

  // conv1 + softmax + centers
  k_conv1  <<<2048, 32, 0, stream>>>(in_h, wb1, fc_b, sm);
  k_centers<<<2048, 256, 0, stream>>>(inputs, sm, featA);

  // graph reasoning iteration 1: featA -> featB
  k_gmsg   <<<32, 256, 0, stream>>>(featA, cwgt_w, Xc, Xc2);
  k_ggemm  <<<64, 32, 0, stream>>>(Xc, wbg, gru_gb, featA, Xc2, Ubuf);
  k_cgemm  <<<32, 32, 0, stream>>>(Xc2, wbc, gru_cb, featA, Ubuf, gamma, featB);
  // iteration 2: featB -> featA
  k_gmsg   <<<32, 256, 0, stream>>>(featB, cwgt_w, Xc, Xc2);
  k_ggemm  <<<64, 32, 0, stream>>>(Xc, wbg, gru_gb, featB, Xc2, Ubuf);
  k_cgemm  <<<32, 32, 0, stream>>>(Xc2, wbc, gru_cb, featB, Ubuf, gamma, featA);

  // collapsed final conv + output assembly
  k_fgemm  <<<144, 32, 0, stream>>>(featA, wb2, Rbuf);
  k_compS  <<<8, 256, 0, stream>>>(Rbuf, conv_b, Sbuf);
  k_out    <<<16384, 256, 0, stream>>>(inputs, Sbuf, out);
}
